// MLPRouter_28312424415695
// MI455X (gfx1250) — matrix operand order repaired and checked by reading, not compiled, not run on hardware
//
#include <hip/hip_runtime.h>

// ---------------------------------------------------------------------------
// MLP router for MI455X (gfx1250, wave32, WMMA bf16 16x16x32, f32 accum)
//   h      = silu(x @ w1^T)      [16384 x 2048]   (bf16 intermediate in ws)
//   logits = h @ w2^T            [16384 x 64]
//   top-8 + softmax per token
// Optimized path: pre-convert x/w1/w2 to bf16 once, then a double-buffered
// LDS GEMM whose staging uses GLOBAL_LOAD_ASYNC_TO_LDS_B128.
// ---------------------------------------------------------------------------

typedef __attribute__((ext_vector_type(16))) __bf16 v16bf;
typedef __attribute__((ext_vector_type(8)))  float  v8f;
typedef __attribute__((ext_vector_type(8)))  __bf16 bf16x8;   // 16 bytes
typedef __attribute__((ext_vector_type(4)))  int    v4i;      // 16 bytes

#define TOKENS 16384
#define HIDDEN 2048
#define NEXP   64
#define TOPK   8
#define LDT    72        // padded halves per LDS row (144 B: 16B-aligned, bank-skewed)

#if defined(__has_builtin)
#if __has_builtin(__builtin_amdgcn_global_load_async_to_lds_b128)
#define HAVE_ASYNC_LDS 1
#endif
#endif

// ---- WMMA fragment loaders (CDNA5 ISA 7.12.2 layouts, wave32) -------------
// A (16x32 bf16, M x K): lane L<16 -> row L, K base 0; lane L>=16 -> row L-16,
// K base 8; halves j<8 -> K=kb+j, j>=8 -> K=kb+16+(j-8).
__device__ __forceinline__ v16bf load_a_frag(const __bf16* base, int lda) {
  const int lane = threadIdx.x & 31;
  const int row  = lane & 15;
  const int kb   = (lane >> 4) << 3;            // 0 or 8
  const __bf16* p = base + row * lda + kb;
  v16bf a;
#pragma unroll
  for (int j = 0; j < 8; ++j) a[j]     = p[j];
#pragma unroll
  for (int j = 0; j < 8; ++j) a[8 + j] = p[16 + j];
  return a;
}

// B (32x16 bf16, K x N) from row-major [N x K] tile: lane L -> col L&15,
// K base 8*(L>>4); halves j<8 -> K=kb+j, j>=8 -> K=kb+16+(j-8).
__device__ __forceinline__ v16bf load_b_frag(const __bf16* base, int ldw) {
  const int lane = threadIdx.x & 31;
  const int col  = lane & 15;
  const int kb   = (lane >> 4) << 3;            // 0 or 8
  const __bf16* p = base + col * ldw + kb;
  v16bf b;
#pragma unroll
  for (int j = 0; j < 8; ++j) b[j]     = p[j];
#pragma unroll
  for (int j = 0; j < 8; ++j) b[8 + j] = p[16 + j];
  return b;
}

// ---- 16-byte global -> LDS copy: async DMA if the toolchain exposes it -----
__device__ __forceinline__ void cp_b128(__bf16* lds, const __bf16* g) {
#if defined(HAVE_ASYNC_LDS)
  __builtin_amdgcn_global_load_async_to_lds_b128(
      (__attribute__((address_space(1))) v4i*)g,
      (__attribute__((address_space(3))) v4i*)lds, 0, 0);
#else
  *(bf16x8*)lds = *(const bf16x8*)g;
#endif
}

__device__ __forceinline__ void wait_cp() {
#if defined(HAVE_ASYNC_LDS)
#if __has_builtin(__builtin_amdgcn_s_wait_asynccnt)
  __builtin_amdgcn_s_wait_asynccnt(0);
#else
  asm volatile("s_wait_asynccnt 0x0" ::: "memory");
#endif
#endif
}

// ---- f32 -> bf16 bulk convert (float4 in, 4x bf16 out) ---------------------
__global__ __launch_bounds__(256)
void cvt_f32_bf16_kernel(const float* __restrict__ src, __bf16* __restrict__ dst,
                         int n4) {
  const int i = blockIdx.x * blockDim.x + threadIdx.x;
  if (i < n4) {
    const float4 v = reinterpret_cast<const float4*>(src)[i];
    __bf16* d = dst + (size_t)i * 4;
    d[0] = (__bf16)v.x; d[1] = (__bf16)v.y; d[2] = (__bf16)v.z; d[3] = (__bf16)v.w;
  }
}

// ---- GEMM1 + SiLU, bf16 inputs, double-buffered LDS, K-step 64 -------------
// Block = 256 threads (8 waves), tile 128(M) x 128(N).
// Wave (wm 0..3, wn 0..1) owns 32x64 = 2x4 accumulators; 16 WMMA / K-step.
__global__ __launch_bounds__(256)
void gemm1_silu_bf16_kernel(const __bf16* __restrict__ xb,
                            const __bf16* __restrict__ w1b,
                            __bf16* __restrict__ h) {
  __shared__ __bf16 sx[2][128 * LDT];   // 18 KB each
  __shared__ __bf16 sw[2][128 * LDT];   // 72 KB total

  const int tid   = threadIdx.x;
  const int lane  = tid & 31;
  const int wid   = tid >> 5;
  const int mbase = blockIdx.y * 128;
  const int nbase = blockIdx.x * 128;
  const int m0    = (wid >> 1) * 32;
  const int n0    = (wid & 1) * 64;

  const v8f vzero = {0.f, 0.f, 0.f, 0.f, 0.f, 0.f, 0.f, 0.f};
  v8f acc[2][4];
#pragma unroll
  for (int i = 0; i < 2; ++i)
#pragma unroll
    for (int j = 0; j < 4; ++j) acc[i][j] = vzero;

  auto stage = [&](int buf, int kt) {
#pragma unroll
    for (int q = 0; q < 4; ++q) {
      const int c  = tid + q * 256;         // 0..1023
      const int r  = c >> 3;                // row 0..127
      const int cc = (c & 7) << 3;          // half-col 0,8,...,56
      cp_b128(&sx[buf][r * LDT + cc], xb  + (size_t)(mbase + r) * HIDDEN + kt + cc);
      cp_b128(&sw[buf][r * LDT + cc], w1b + (size_t)(nbase + r) * HIDDEN + kt + cc);
    }
  };

  stage(0, 0);
  wait_cp();
  __syncthreads();

  int buf = 0;
  for (int kt = 0; kt < HIDDEN; kt += 64) {
    if (kt + 64 < HIDDEN) stage(buf ^ 1, kt + 64);   // overlap DMA with WMMAs

#pragma unroll
    for (int kk = 0; kk < 64; kk += 32) {
      const v16bf a0 = load_a_frag(&sx[buf][(m0 +  0) * LDT + kk], LDT);
      const v16bf a1 = load_a_frag(&sx[buf][(m0 + 16) * LDT + kk], LDT);
#pragma unroll
      for (int j = 0; j < 4; ++j) {
        const v16bf b = load_b_frag(&sw[buf][(n0 + 16 * j) * LDT + kk], LDT);
        acc[0][j] = __builtin_amdgcn_wmma_f32_16x16x32_bf16(
            false, a0, false, b, (short)0, acc[0][j], false, false);
        acc[1][j] = __builtin_amdgcn_wmma_f32_16x16x32_bf16(
            false, a1, false, b, (short)0, acc[1][j], false, false);
      }
    }

    wait_cp();
    __syncthreads();
    buf ^= 1;
  }

  // Epilogue: SiLU, store bf16. C layout: VGPR e -> row e + 8*(lane>=16), col=lane&15.
  const int rlo = (lane >> 4) << 3;
  const int col = lane & 15;
#pragma unroll
  for (int i = 0; i < 2; ++i)
#pragma unroll
    for (int j = 0; j < 4; ++j)
#pragma unroll
      for (int e = 0; e < 8; ++e) {
        const int gr = mbase + m0 + 16 * i + rlo + e;
        const int gc = nbase + n0 + 16 * j + col;
        const float v = acc[i][j][e];
        const float s = v / (1.0f + __expf(-v));
        h[(size_t)gr * HIDDEN + gc] = (__bf16)s;
      }
}

// ---- Fallback GEMM1 (f32 inputs staged+converted in LDS) -------------------
__global__ __launch_bounds__(256)
void gemm1_silu_f32_kernel(const float* __restrict__ x, const float* __restrict__ w1,
                           __bf16* __restrict__ h) {
  __shared__ __bf16 sx[128 * 40];
  __shared__ __bf16 sw[128 * 40];

  const int tid   = threadIdx.x;
  const int lane  = tid & 31;
  const int wid   = tid >> 5;
  const int mbase = blockIdx.y * 128;
  const int nbase = blockIdx.x * 128;
  const int m0    = (wid >> 1) * 32;
  const int n0    = (wid & 1) * 64;

  const v8f vzero = {0.f, 0.f, 0.f, 0.f, 0.f, 0.f, 0.f, 0.f};
  v8f acc[2][4];
#pragma unroll
  for (int i = 0; i < 2; ++i)
#pragma unroll
    for (int j = 0; j < 4; ++j) acc[i][j] = vzero;

  for (int kt = 0; kt < HIDDEN; kt += 32) {
#pragma unroll
    for (int q = 0; q < 4; ++q) {
      const int lin = tid + q * 256;
      const int r   = lin >> 3;
      const int c4  = (lin & 7) << 2;
      const float4 vx = *reinterpret_cast<const float4*>(
          &x[(size_t)(mbase + r) * HIDDEN + kt + c4]);
      __bf16* dx = &sx[r * 40 + c4];
      dx[0] = (__bf16)vx.x; dx[1] = (__bf16)vx.y;
      dx[2] = (__bf16)vx.z; dx[3] = (__bf16)vx.w;
      const float4 vw = *reinterpret_cast<const float4*>(
          &w1[(size_t)(nbase + r) * HIDDEN + kt + c4]);
      __bf16* dw = &sw[r * 40 + c4];
      dw[0] = (__bf16)vw.x; dw[1] = (__bf16)vw.y;
      dw[2] = (__bf16)vw.z; dw[3] = (__bf16)vw.w;
    }
    __syncthreads();

    const v16bf a0 = load_a_frag(&sx[(m0 +  0) * 40], 40);
    const v16bf a1 = load_a_frag(&sx[(m0 + 16) * 40], 40);
#pragma unroll
    for (int j = 0; j < 4; ++j) {
      const v16bf b = load_b_frag(&sw[(n0 + 16 * j) * 40], 40);
      acc[0][j] = __builtin_amdgcn_wmma_f32_16x16x32_bf16(
          false, a0, false, b, (short)0, acc[0][j], false, false);
      acc[1][j] = __builtin_amdgcn_wmma_f32_16x16x32_bf16(
          false, a1, false, b, (short)0, acc[1][j], false, false);
    }
    __syncthreads();
  }

  const int rlo = (lane >> 4) << 3;
  const int col = lane & 15;
#pragma unroll
  for (int i = 0; i < 2; ++i)
#pragma unroll
    for (int j = 0; j < 4; ++j)
#pragma unroll
      for (int e = 0; e < 8; ++e) {
        const int gr = mbase + m0 + 16 * i + rlo + e;
        const int gc = nbase + n0 + 16 * j + col;
        const float v = acc[i][j][e];
        const float s = v / (1.0f + __expf(-v));
        h[(size_t)gr * HIDDEN + gc] = (__bf16)s;
      }
}

// ---- GEMM2 + top-8 + softmax -----------------------------------------------
// Block = 128 threads (4 waves); each wave: 16 tokens x 64 experts (4 WMMA accs).
__global__ __launch_bounds__(128)
void gemm2_router_kernel(const __bf16* __restrict__ h, const __bf16* __restrict__ w2b,
                         float* __restrict__ logits, float* __restrict__ weights,
                         int* __restrict__ sel) {
  __shared__ float slog[4][16][66];

  const int tid     = threadIdx.x;
  const int lane    = tid & 31;
  const int wv      = tid >> 5;
  const int tokbase = blockIdx.x * 64 + wv * 16;

  const v8f vzero = {0.f, 0.f, 0.f, 0.f, 0.f, 0.f, 0.f, 0.f};
  v8f acc[4];
#pragma unroll
  for (int j = 0; j < 4; ++j) acc[j] = vzero;

  for (int kt = 0; kt < HIDDEN; kt += 32) {
    const v16bf a = load_a_frag(h + (size_t)tokbase * HIDDEN + kt, HIDDEN);
#pragma unroll
    for (int j = 0; j < 4; ++j) {
      const v16bf b = load_b_frag(w2b + (size_t)(16 * j) * HIDDEN + kt, HIDDEN);
      acc[j] = __builtin_amdgcn_wmma_f32_16x16x32_bf16(
          false, a, false, b, (short)0, acc[j], false, false);
    }
  }

  const int rlo = (lane >> 4) << 3;
  const int col = lane & 15;
#pragma unroll
  for (int j = 0; j < 4; ++j)
#pragma unroll
    for (int e = 0; e < 8; ++e) {
      const int t  = rlo + e;
      const int ex = 16 * j + col;
      const float v = acc[j][e];
      logits[(size_t)(tokbase + t) * NEXP + ex] = v;
      slog[wv][t][ex] = v;
    }
  __syncthreads();

  if (lane < 16) {
    const int t = tokbase + lane;
    float best[TOPK];
    int   bidx[TOPK];
#pragma unroll
    for (int i = 0; i < TOPK; ++i) { best[i] = -3.4e38f; bidx[i] = 0; }

    for (int e = 0; e < NEXP; ++e) {
      const float v = slog[wv][lane][e];
      if (v > best[TOPK - 1]) {         // strict '>' keeps lowest index on ties
        best[TOPK - 1] = v;
        bidx[TOPK - 1] = e;
#pragma unroll
        for (int i = TOPK - 1; i > 0; --i) {
          if (best[i] > best[i - 1]) {
            const float tv = best[i]; best[i] = best[i - 1]; best[i - 1] = tv;
            const int   ti = bidx[i]; bidx[i] = bidx[i - 1]; bidx[i - 1] = ti;
          }
        }
      }
    }

    float s = 0.0f, w[TOPK];
#pragma unroll
    for (int i = 0; i < TOPK; ++i) { w[i] = __expf(best[i] - best[0]); s += w[i]; }
    const float inv = 1.0f / s;
#pragma unroll
    for (int i = 0; i < TOPK; ++i) {
      weights[(size_t)t * TOPK + i] = w[i] * inv;
      sel[(size_t)t * TOPK + i]     = bidx[i];
    }
  }
}

// ---------------------------------------------------------------------------
extern "C" void kernel_launch(void* const* d_in, const int* in_sizes, int n_in,
                              void* d_out, int out_size, void* d_ws, size_t ws_size,
                              hipStream_t stream) {
  const float* x  = (const float*)d_in[0];   // [16384, 2048]
  const float* w1 = (const float*)d_in[1];   // [2048, 2048]
  const float* w2 = (const float*)d_in[2];   // [64, 2048]

  float* out       = (float*)d_out;
  float* d_weights = out;                                    // [T, 8] f32
  int*   d_sel     = (int*)(out + (size_t)TOKENS * TOPK);    // [T, 8] i32
  float* d_logits  = out + (size_t)TOKENS * TOPK * 2;        // [T, 64] f32

  // Workspace layout (bytes):
  const size_t off_h   = 0;                                   // 64 MB bf16 h
  const size_t off_w2b = off_h   + (size_t)TOKENS * HIDDEN * 2;
  const size_t off_xb  = off_w2b + (size_t)NEXP   * HIDDEN * 2;
  const size_t off_w1b = off_xb  + (size_t)TOKENS * HIDDEN * 2;
  const size_t need    = off_w1b + (size_t)HIDDEN * HIDDEN * 2;

  __bf16* h   = (__bf16*)((char*)d_ws + off_h);
  __bf16* w2b = (__bf16*)((char*)d_ws + off_w2b);

  hipLaunchKernelGGL(cvt_f32_bf16_kernel, dim3((NEXP * HIDDEN / 4 + 255) / 256),
                     dim3(256), 0, stream, w2, w2b, NEXP * HIDDEN / 4);

  if (ws_size >= need) {
    __bf16* xb  = (__bf16*)((char*)d_ws + off_xb);
    __bf16* w1b = (__bf16*)((char*)d_ws + off_w1b);
    hipLaunchKernelGGL(cvt_f32_bf16_kernel, dim3(TOKENS * HIDDEN / 4 / 256),
                       dim3(256), 0, stream, x, xb, TOKENS * HIDDEN / 4);
    hipLaunchKernelGGL(cvt_f32_bf16_kernel, dim3(HIDDEN * HIDDEN / 4 / 256),
                       dim3(256), 0, stream, w1, w1b, HIDDEN * HIDDEN / 4);
    hipLaunchKernelGGL(gemm1_silu_bf16_kernel, dim3(HIDDEN / 128, TOKENS / 128),
                       dim3(256), 0, stream, xb, w1b, h);
  } else {
    hipLaunchKernelGGL(gemm1_silu_f32_kernel, dim3(HIDDEN / 128, TOKENS / 128),
                       dim3(256), 0, stream, x, w1, h);
  }

  hipLaunchKernelGGL(gemm2_router_kernel, dim3(TOKENS / 64), dim3(128), 0, stream,
                     h, w2b, d_logits, d_weights, d_sel);
}
